// MPLayer_88218628260532
// MI455X (gfx1250) — matrix operand order repaired and checked by reading, not compiled, not run on hardware
//
#include <hip/hip_runtime.h>

// ---- CDNA5 WMMA types -------------------------------------------------------
typedef __attribute__((ext_vector_type(16))) _Float16 v16h;
typedef __attribute__((ext_vector_type(8)))  _Float16 v8h;
typedef __attribute__((ext_vector_type(8)))  float    v8f;

union AFrag { v16h v; v8h h[2]; };

__device__ __forceinline__ float lane_shfl(float v, int srcLane) {
    // wave32 lane gather via LDS-permute hardware (no LDS memory used)
    return __int_as_float(__builtin_amdgcn_ds_bpermute(srcLane << 2, __float_as_int(v)));
}

static constexpr int BB = 4, NN = 20000, KK = 16, DD = 64, OO = 64;
static constexpr int BN = BB * NN;          // 80000 nodes, divisible by 16
// workspace layout (bytes)
static constexpr size_t WS_BPACK = 0;                     // 32 combos * 32 lanes * 32B = 32 KB
static constexpr size_t WS_XH    = 32768;                 // f16 x copy: BN*64*2 = 10,240,000
static constexpr size_t WS_S     = 32768 + 10240000;      // f32 self term: BN*64*4 = 20,480,000
static constexpr size_t WS_MSG   = 32768 + 10240000 + 20480000; // f16 messages: 10,240,000

// ---- Pack B operands (We_self, We_nbr, Wn) into WMMA wave32 B layout --------
// B 32x16 f16: lane L holds column o = 16t + (L&15); j<8 -> K = 8h+j, j>=8 -> K = 16+8h+(j-8), h = L>>4.
// combo 0..7  : We_self  (c=combo/4, t=combo%4)
// combo 8..15 : We_nbr
// combo 16..31: Wn (K=128 -> c=0..3)
__global__ void k_pack_b(const float* __restrict__ We, const float* __restrict__ Wn,
                         _Float16* __restrict__ Bp) {
    int u = blockIdx.x * 256 + threadIdx.x;        // 0..1023
    int combo = u >> 5, lane = u & 31;
    const float* W; int rowoff, c, t;
    if (combo < 8)       { W = We; rowoff = 0;  c = combo >> 2;        t = combo & 3; }
    else if (combo < 16) { W = We; rowoff = 64; c = (combo - 8) >> 2;  t = combo & 3; }
    else                 { W = Wn; rowoff = 0;  c = (combo - 16) >> 2; t = combo & 3; }
    int o = t * 16 + (lane & 15);
    _Float16* dst = Bp + (size_t)(combo * 32 + lane) * 16;
    #pragma unroll
    for (int j = 0; j < 16; ++j) {
        int Kv = (j < 8) ? ((lane >> 4) * 8 + j) : (16 + (lane >> 4) * 8 + (j - 8));
        dst[j] = (_Float16)W[(size_t)(rowoff + c * 32 + Kv) * 64 + o];
    }
}

// ---- Convert x to f16 (halves random-gather L2 traffic; A operand is f16) ---
__global__ void k_cvt_xh(const float* __restrict__ x, _Float16* __restrict__ xh) {
    int i = blockIdx.x * 256 + threadIdx.x;
    if (i < BN * 64) xh[i] = (_Float16)x[i];
}

// ---- S = x @ We_self + be   (16 nodes per wave, 8 WMMAs) --------------------
__global__ void k_self(const _Float16* __restrict__ xh, const _Float16* __restrict__ Bp,
                       const float* __restrict__ be, float* __restrict__ S) {
    int lane = threadIdx.x & 31, wv = threadIdx.x >> 5;
    int l16 = lane & 15, hf = lane >> 4;
    int g0 = (blockIdx.x * 8 + wv) * 16;
    // A 16x32 layout: row M = l16; K runs [32c + 8*hf, +7] and [+16, +23] -> aligned b128s
    const v8h* arow = (const v8h*)(xh + (size_t)(g0 + l16) * 64);
    AFrag a[2];
    a[0].h[0] = arow[hf];     a[0].h[1] = arow[hf + 2];
    a[1].h[0] = arow[4 + hf]; a[1].h[1] = arow[6 + hf];
    v8f acc[4] = {};
    #pragma unroll
    for (int t = 0; t < 4; ++t)
        #pragma unroll
        for (int c = 0; c < 2; ++c) {
            const v8h* bp = (const v8h*)(Bp + (size_t)((c * 4 + t) * 32 + lane) * 16);
            AFrag bm; bm.h[0] = bp[0]; bm.h[1] = bp[1];
            acc[t] = __builtin_amdgcn_wmma_f32_16x16x32_f16(false, a[c].v, false, bm.v,
                                                            (short)0, acc[t], false, false);
        }
    #pragma unroll
    for (int t = 0; t < 4; ++t) {
        int o = t * 16 + l16;
        float bias = be[o];
        #pragma unroll
        for (int r = 0; r < 8; ++r)                 // C/D: row M = r + 8*hf, col = l16
            S[(size_t)(g0 + r + hf * 8) * 64 + o] = acc[t][r] + bias;
    }
}

// ---- Edge kernel: one wave per node -----------------------------------------
// A rows = 16 gathered neighbors (direct f16 gather, 2x b128 per K-chunk per lane),
// 8 WMMAs; rel-pos term in VALU; S broadcast; ReLU + mean over K via ds_bpermute.
__global__ void k_edge(const _Float16* __restrict__ xh, const _Float16* __restrict__ Bp,
                       const long long* __restrict__ eidx, const float* __restrict__ pos,
                       const float* __restrict__ We, const float* __restrict__ S,
                       _Float16* __restrict__ msg) {
    int lane = threadIdx.x & 31, wv = threadIdx.x >> 5;
    int l16 = lane & 15, hf = lane >> 4;
    int g = blockIdx.x * 8 + wv;                    // node id, grid sized exactly
    int b = g / NN;
    long long e = eidx[(size_t)g * KK + l16];       // neighbor for row M = l16
    size_t ng = (size_t)b * NN + (size_t)e;
    float rx = pos[ng * 2 + 0] - pos[(size_t)g * 2 + 0];
    float ry = pos[ng * 2 + 1] - pos[(size_t)g * 2 + 1];
    const v8h* arow = (const v8h*)(xh + ng * 64);
    AFrag a[2];
    a[0].h[0] = arow[hf];     a[0].h[1] = arow[hf + 2];
    a[1].h[0] = arow[4 + hf]; a[1].h[1] = arow[6 + hf];
    // rel-pos of the 8 rows this lane-half accumulates
    float rxr[8], ryr[8];
    #pragma unroll
    for (int r = 0; r < 8; ++r) {
        int src = hf * 8 + r;
        rxr[r] = lane_shfl(rx, src);
        ryr[r] = lane_shfl(ry, src);
    }
    v8f acc[4] = {};
    #pragma unroll
    for (int t = 0; t < 4; ++t)
        #pragma unroll
        for (int c = 0; c < 2; ++c) {
            const v8h* bp = (const v8h*)(Bp + (size_t)((8 + c * 4 + t) * 32 + lane) * 16);
            AFrag bm; bm.h[0] = bp[0]; bm.h[1] = bp[1];
            acc[t] = __builtin_amdgcn_wmma_f32_16x16x32_f16(false, a[c].v, false, bm.v,
                                                            (short)0, acc[t], false, false);
        }
    #pragma unroll
    for (int t = 0; t < 4; ++t) {
        int o = t * 16 + l16;
        float sv = S[(size_t)g * 64 + o];           // self term + be, same for all k
        float c0 = We[(size_t)(2 * DD + 0) * 64 + o];
        float c1 = We[(size_t)(2 * DD + 1) * 64 + o];
        float sum = 0.f;
        #pragma unroll
        for (int r = 0; r < 8; ++r) {
            float v = acc[t][r] + sv + rxr[r] * c0 + ryr[r] * c1;
            sum += fmaxf(v, 0.f);
        }
        sum += lane_shfl(sum, lane ^ 16);           // combine the two k-halves
        if (hf == 0) msg[(size_t)g * 64 + o] = (_Float16)(sum * (1.0f / 16.0f));
    }
}

// ---- Node kernel: h = [x||msg] @ Wn + bn, LayerNorm, ReLU, mask -------------
__global__ void k_node(const _Float16* __restrict__ xh, const _Float16* __restrict__ msg,
                       const _Float16* __restrict__ Bp, const float* __restrict__ bn,
                       const float* __restrict__ gamma, const float* __restrict__ beta,
                       const float* __restrict__ mask, float* __restrict__ out) {
    int lane = threadIdx.x & 31, wv = threadIdx.x >> 5;
    int l16 = lane & 15, hf = lane >> 4;
    int g0 = (blockIdx.x * 8 + wv) * 16;
    const v8h* ax = (const v8h*)(xh  + (size_t)(g0 + l16) * 64);
    const v8h* am = (const v8h*)(msg + (size_t)(g0 + l16) * 64);
    v8f acc[4] = {};
    #pragma unroll
    for (int c = 0; c < 4; ++c) {                   // K = 128: x then messages
        const v8h* src = (c < 2) ? ax : am;
        int cc = c & 1;
        AFrag a;
        a.h[0] = src[cc * 4 + hf];
        a.h[1] = src[cc * 4 + 2 + hf];
        #pragma unroll
        for (int t = 0; t < 4; ++t) {
            const v8h* bp = (const v8h*)(Bp + (size_t)((16 + c * 4 + t) * 32 + lane) * 16);
            AFrag bm; bm.h[0] = bp[0]; bm.h[1] = bp[1];
            acc[t] = __builtin_amdgcn_wmma_f32_16x16x32_f16(false, a.v, false, bm.v,
                                                            (short)0, acc[t], false, false);
        }
    }
    float go[4], bo[4];
    #pragma unroll
    for (int t = 0; t < 4; ++t) {
        int o = t * 16 + l16;
        float bb = bn[o];
        #pragma unroll
        for (int r = 0; r < 8; ++r) acc[t][r] += bb;
        go[t] = gamma[o]; bo[t] = beta[o];
    }
    // LayerNorm: each row (node) lives entirely in one 16-lane half (4 values/lane)
    #pragma unroll
    for (int r = 0; r < 8; ++r) {
        float s = acc[0][r] + acc[1][r] + acc[2][r] + acc[3][r];
        float q = acc[0][r] * acc[0][r] + acc[1][r] * acc[1][r]
                + acc[2][r] * acc[2][r] + acc[3][r] * acc[3][r];
        #pragma unroll
        for (int off = 1; off < 16; off <<= 1) {
            s += lane_shfl(s, lane ^ off);
            q += lane_shfl(q, lane ^ off);
        }
        float mean = s * (1.0f / 64.0f);
        float var  = q * (1.0f / 64.0f) - mean * mean;
        float inv  = rsqrtf(var + 1e-5f);
        int row = g0 + r + hf * 8;
        float mk = mask[row];
        #pragma unroll
        for (int t = 0; t < 4; ++t) {
            int o = t * 16 + l16;
            float v = (acc[t][r] - mean) * inv * go[t] + bo[t];
            out[(size_t)row * 64 + o] = fmaxf(v, 0.f) * mk;
        }
    }
}

// ---- Launcher ---------------------------------------------------------------
extern "C" void kernel_launch(void* const* d_in, const int* in_sizes, int n_in,
                              void* d_out, int out_size, void* d_ws, size_t ws_size,
                              hipStream_t stream) {
    const float*     x    = (const float*)d_in[0];
    const float*     pos  = (const float*)d_in[1];
    const long long* eidx = (const long long*)d_in[2];   // int64 in reference
    const float*     mask = (const float*)d_in[3];
    const float*     We   = (const float*)d_in[4];
    const float*     be   = (const float*)d_in[5];
    const float*     Wn   = (const float*)d_in[6];
    const float*     bn   = (const float*)d_in[7];
    const float*     gam  = (const float*)d_in[8];
    const float*     bet  = (const float*)d_in[9];
    float* out = (float*)d_out;

    char* w = (char*)d_ws;
    _Float16* Bp  = (_Float16*)(w + WS_BPACK);
    _Float16* xh  = (_Float16*)(w + WS_XH);
    float*    S   = (float*)   (w + WS_S);
    _Float16* msg = (_Float16*)(w + WS_MSG);

    k_pack_b<<<4, 256, 0, stream>>>(We, Wn, Bp);
    k_cvt_xh<<<(BN * 64 + 255) / 256, 256, 0, stream>>>(x, xh);
    k_self<<<BN / 16 / 8, 256, 0, stream>>>(xh, Bp, be, S);        // 625 blocks
    k_edge<<<BN / 8, 256, 0, stream>>>(xh, Bp, eidx, pos, We, S, msg); // 10000 blocks
    k_node<<<BN / 16 / 8, 256, 0, stream>>>(xh, msg, Bp, bn, gam, bet, mask, out);
}
